// ConditionalFeedForward_63711544869416
// MI455X (gfx1250) — matrix operand order repaired and checked by reading, not compiled, not run on hardware
//
#include <hip/hip_runtime.h>

// ---------------------------------------------------------------------------
// MoE ConditionalFeedForward (SwiGLU, top-2 of 8 experts) for MI455X / gfx1250
//
// Fast path: bf16 hi/lo split-precision GEMMs on V_WMMA_F32_16X16X32_BF16.
//   a ≈ a_hi + a_lo (bf16 each);  a*b ≈ a_lo*b_hi + a_hi*b_lo + a_hi*b_hi
//   -> 3 bf16 WMMAs replace 8 fp32 16x16x4 WMMAs per K=32 (2.7x fewer XDL
//      issue slots), fp32 accumulation, ~16-bit effective mantissa.
// Weights/x are pre-split into bf16 planes once per launch (w2 additionally
// transposed to [e][d][h]) so GEMM staging is pure b128 copies and every
// WMMA fragment is a contiguous LDS run (k = 16*half + s, same order for A and B).
// Fallback (small ws): fp32 V_WMMA_F32_16X16X4_F32 path.
// ---------------------------------------------------------------------------

namespace {
constexpr int TT    = 2048;           // tokens
constexpr int DD    = 1024;           // model dim
constexpr int HH    = 2816;           // hidden dim
constexpr int EE    = 8;              // experts
constexpr int PAIRS = TT * 2;         // 4096 (token, slot) pairs
}

typedef unsigned short u16;
typedef float v2f  __attribute__((ext_vector_type(2)));
typedef float v8f  __attribute__((ext_vector_type(8)));
typedef __bf16 v16bf __attribute__((ext_vector_type(16)));

// ---------------- bf16 helpers -------------------------------------------
__device__ __forceinline__ u16 f2bf(float f) {           // round-to-nearest-even
  unsigned u = __float_as_uint(f);
  unsigned r = u + 0x7FFFu + ((u >> 16) & 1u);
  return (u16)(r >> 16);
}
__device__ __forceinline__ float bf2f(u16 h) {
  return __uint_as_float(((unsigned)h) << 16);
}

// ---------------- WMMA wrappers ------------------------------------------
__device__ __forceinline__ v8f wmma_bf(v16bf a, v16bf b, v8f c) {
  return __builtin_amdgcn_wmma_f32_16x16x32_bf16(
      false, a, false, b, (short)0, c, false, false);
}
__device__ __forceinline__ v8f wmma_f32(v2f a, v2f b, v8f c) {
  return __builtin_amdgcn_wmma_f32_16x16x4_f32(
      false, a, false, b, (short)0, c, false, false);
}

// A frag 16x32 bf16: lane half h: elems = A[m][kk+16h .. +15] (contiguous 32B)
__device__ __forceinline__ v16bf ld_fragA(const u16* row, int kk, int half8) {
  union { v16bf v; uint4 q[2]; } u;
  u.q[0] = *(const uint4*)(row + kk + 2 * half8);
  u.q[1] = *(const uint4*)(row + kk + 2 * half8 + 8);
  return u.v;
}
// B frag 32x16 bf16: lane half h: elems = B[n][kk+16h .. +15] (contiguous 32B)
__device__ __forceinline__ v16bf ld_fragB(const u16* row, int kk, int half16) {
  union { v16bf v; uint4 q[2]; } u;
  u.q[0] = *(const uint4*)(row + kk + half16);
  u.q[1] = *(const uint4*)(row + kk + half16 + 8);
  return u.v;
}

// ---------------------------------------------------------------------------
// Phase 0: route pairs into per-expert lists
// ---------------------------------------------------------------------------
__global__ void moe_route(const int* __restrict__ idx,
                          int* __restrict__ counts,
                          int* __restrict__ lists) {
  int p = blockIdx.x * blockDim.x + threadIdx.x;
  if (p >= PAIRS) return;
  int e = idx[p];
  int slot = atomicAdd(&counts[e], 1);
  lists[e * PAIRS + slot] = p;
}

// ---------------------------------------------------------------------------
// fp32 -> bf16 hi/lo split (row-major preserving), float4 granular
// ---------------------------------------------------------------------------
__global__ void conv_split(const float* __restrict__ src,
                           u16* __restrict__ hi, u16* __restrict__ lo,
                           long n4) {
  long i = (long)blockIdx.x * blockDim.x + threadIdx.x;
  if (i >= n4) return;
  float4 v = ((const float4*)src)[i];
  ushort4 h, l;
  h.x = f2bf(v.x); l.x = f2bf(v.x - bf2f(h.x));
  h.y = f2bf(v.y); l.y = f2bf(v.y - bf2f(h.y));
  h.z = f2bf(v.z); l.z = f2bf(v.z - bf2f(h.z));
  h.w = f2bf(v.w); l.w = f2bf(v.w - bf2f(h.w));
  ((ushort4*)hi)[i] = h;
  ((ushort4*)lo)[i] = l;
}

// ---------------------------------------------------------------------------
// w2 [e][h][d] fp32  ->  transposed bf16 hi/lo planes [e][d][h]
// 32x32 LDS tile transpose, 256 threads
// ---------------------------------------------------------------------------
__global__ __launch_bounds__(256)
void conv_w2t(const float* __restrict__ w2,
              u16* __restrict__ hi, u16* __restrict__ lo) {
  const int e  = blockIdx.z;
  const int h0 = blockIdx.y * 32;
  const int d0 = blockIdx.x * 32;
  __shared__ float sT[32][33];

  const int tid = threadIdx.x;
  {
    int r  = tid >> 3;           // h row in tile
    int c4 = (tid & 7) << 2;     // d col in tile
    float4 v = *(const float4*)(w2 + (size_t)(e * HH + h0 + r) * DD + d0 + c4);
    sT[r][c4 + 0] = v.x; sT[r][c4 + 1] = v.y;
    sT[r][c4 + 2] = v.z; sT[r][c4 + 3] = v.w;
  }
  __syncthreads();
  {
    int dr   = tid >> 3;         // d row in tile
    int hseg = (tid & 7) << 2;   // h col in tile
    ushort4 hq, lq;
    float f0 = sT[hseg + 0][dr]; hq.x = f2bf(f0); lq.x = f2bf(f0 - bf2f(hq.x));
    float f1 = sT[hseg + 1][dr]; hq.y = f2bf(f1); lq.y = f2bf(f1 - bf2f(hq.y));
    float f2 = sT[hseg + 2][dr]; hq.z = f2bf(f2); lq.z = f2bf(f2 - bf2f(hq.z));
    float f3 = sT[hseg + 3][dr]; hq.w = f2bf(f3); lq.w = f2bf(f3 - bf2f(hq.w));
    size_t o = (size_t)(e * DD + d0 + dr) * HH + h0 + hseg;
    *(ushort4*)(hi + o) = hq;
    *(ushort4*)(lo + o) = lq;
  }
}

// ---------------------------------------------------------------------------
// Phase 1 (bf16x3): g[p,h] = silu(x·w1^T) * (x·w3^T); g written as hi/lo bf16
// block = 256 thr (8 waves), tile = 32 pairs x 128 h, K-chunk = 64 over D
// ---------------------------------------------------------------------------
__global__ __launch_bounds__(256)
void moe_gemm1_bf(const u16* __restrict__ xh,  const u16* __restrict__ xl,
                  const u16* __restrict__ w1h, const u16* __restrict__ w1l,
                  const u16* __restrict__ w3h, const u16* __restrict__ w3l,
                  const int* __restrict__ counts, const int* __restrict__ lists,
                  u16* __restrict__ gh, u16* __restrict__ gl) {
  const int e    = blockIdx.z;
  const int cnt  = counts[e];
  const int row0 = blockIdx.y * 32;
  if (row0 >= cnt) return;                     // uniform early exit
  const int h0   = blockIdx.x * 128;

  __shared__ __align__(16) u16 sAh[32][72],  sAl[32][72];
  __shared__ __align__(16) u16 sB1h[128][72], sB1l[128][72];
  __shared__ __align__(16) u16 sB3h[128][72], sB3l[128][72];
  __shared__ int sPair[32];

  const int tid  = threadIdx.x;
  const int wave = tid >> 5;
  const int lane = tid & 31;
  const int half = lane >> 4;
  const int mrow = lane & 15;
  const int nn   = (wave << 4) + (lane & 15);
  const int aoff = half << 3;                  // 8*half
  const int boff = half << 4;                  // 16*half

  if (tid < 32) {
    int r = row0 + tid;
    sPair[tid] = (r < cnt) ? lists[e * PAIRS + r] : -1;
  }
  __syncthreads();

  v8f acc10 = {}, acc11 = {}, acc30 = {}, acc31 = {};

  for (int kc = 0; kc < DD; kc += 64) {
    __syncthreads();
    // stage A planes: 32 rows x 64 bf16 = 256 uint4 per plane (1/thread)
    {
      int r = tid >> 3;
      int c = (tid & 7) << 3;
      int p = sPair[r];
      uint4 vh = make_uint4(0, 0, 0, 0), vl = vh;
      if (p >= 0) {
        size_t o = (size_t)(p >> 1) * DD + kc + c;
        vh = *(const uint4*)(xh + o);
        vl = *(const uint4*)(xl + o);
      }
      *(uint4*)(&sAh[r][c]) = vh;
      *(uint4*)(&sAl[r][c]) = vl;
    }
    // stage B planes: 128 rows x 64 bf16 = 1024 uint4 per plane (4/thread)
#pragma unroll
    for (int i = 0; i < 4; ++i) {
      int lin = tid + (i << 8);
      int n   = lin >> 3;
      int c   = (lin & 7) << 3;
      size_t o = (size_t)(e * HH + h0 + n) * DD + kc + c;
      *(uint4*)(&sB1h[n][c]) = *(const uint4*)(w1h + o);
      *(uint4*)(&sB1l[n][c]) = *(const uint4*)(w1l + o);
      *(uint4*)(&sB3h[n][c]) = *(const uint4*)(w3h + o);
      *(uint4*)(&sB3l[n][c]) = *(const uint4*)(w3l + o);
    }
    __syncthreads();

#pragma unroll
    for (int kk = 0; kk < 64; kk += 32) {
      v16bf a0h = ld_fragA(&sAh[mrow][0],      kk, aoff);
      v16bf a0l = ld_fragA(&sAl[mrow][0],      kk, aoff);
      v16bf a1h = ld_fragA(&sAh[16 + mrow][0], kk, aoff);
      v16bf a1l = ld_fragA(&sAl[16 + mrow][0], kk, aoff);
      v16bf b1h = ld_fragB(&sB1h[nn][0], kk, boff);
      v16bf b1l = ld_fragB(&sB1l[nn][0], kk, boff);
      v16bf b3h = ld_fragB(&sB3h[nn][0], kk, boff);
      v16bf b3l = ld_fragB(&sB3l[nn][0], kk, boff);
      acc10 = wmma_bf(a0l, b1h, acc10);
      acc10 = wmma_bf(a0h, b1l, acc10);
      acc10 = wmma_bf(a0h, b1h, acc10);
      acc11 = wmma_bf(a1l, b1h, acc11);
      acc11 = wmma_bf(a1h, b1l, acc11);
      acc11 = wmma_bf(a1h, b1h, acc11);
      acc30 = wmma_bf(a0l, b3h, acc30);
      acc30 = wmma_bf(a0h, b3l, acc30);
      acc30 = wmma_bf(a0h, b3h, acc30);
      acc31 = wmma_bf(a1l, b3h, acc31);
      acc31 = wmma_bf(a1h, b3l, acc31);
      acc31 = wmma_bf(a1h, b3h, acc31);
    }
  }

  // fused SwiGLU epilogue; C/D layout: row = vi + 8*half (+16 for tile 1)
  const int mbase = half << 3;
#pragma unroll
  for (int vi = 0; vi < 8; ++vi) {
    int mo0 = vi + mbase;
    int p0  = sPair[mo0];
    if (p0 >= 0) {
      float a1v = acc10[vi], a3v = acc30[vi];
      float gv  = (a1v / (1.0f + __expf(-a1v))) * a3v;
      size_t o  = (size_t)p0 * HH + h0 + nn;
      u16 h = f2bf(gv);
      gh[o] = h; gl[o] = f2bf(gv - bf2f(h));
    }
    int mo1 = 16 + vi + mbase;
    int p1  = sPair[mo1];
    if (p1 >= 0) {
      float a1v = acc11[vi], a3v = acc31[vi];
      float gv  = (a1v / (1.0f + __expf(-a1v))) * a3v;
      size_t o  = (size_t)p1 * HH + h0 + nn;
      u16 h = f2bf(gv);
      gh[o] = h; gl[o] = f2bf(gv - bf2f(h));
    }
  }
}

// ---------------------------------------------------------------------------
// Phase 2 (bf16x3): out[p,d] = g[p,:] @ w2[e]  (w2 pre-transposed to [d][h])
// block = 256 thr (8 waves), tile = 32 pairs x 128 d, K-chunk = 64 over H
// ---------------------------------------------------------------------------
__global__ __launch_bounds__(256)
void moe_gemm2_bf(const u16* __restrict__ gh,  const u16* __restrict__ gl,
                  const u16* __restrict__ w2h, const u16* __restrict__ w2l,
                  const int* __restrict__ counts, const int* __restrict__ lists,
                  float* __restrict__ out) {
  const int e    = blockIdx.z;
  const int cnt  = counts[e];
  const int row0 = blockIdx.y * 32;
  if (row0 >= cnt) return;
  const int d0   = blockIdx.x * 128;

  __shared__ __align__(16) u16 sAh[32][72],  sAl[32][72];
  __shared__ __align__(16) u16 sBh[128][72], sBl[128][72];
  __shared__ int sPair[32];

  const int tid  = threadIdx.x;
  const int wave = tid >> 5;
  const int lane = tid & 31;
  const int half = lane >> 4;
  const int mrow = lane & 15;
  const int nn   = (wave << 4) + (lane & 15);
  const int aoff = half << 3;
  const int boff = half << 4;

  if (tid < 32) {
    int r = row0 + tid;
    sPair[tid] = (r < cnt) ? lists[e * PAIRS + r] : -1;
  }
  __syncthreads();

  v8f acc0 = {}, acc1 = {};

  for (int kc = 0; kc < HH; kc += 64) {
    __syncthreads();
    {
      int r = tid >> 3;
      int c = (tid & 7) << 3;
      int p = sPair[r];
      uint4 vh = make_uint4(0, 0, 0, 0), vl = vh;
      if (p >= 0) {
        size_t o = (size_t)p * HH + kc + c;
        vh = *(const uint4*)(gh + o);
        vl = *(const uint4*)(gl + o);
      }
      *(uint4*)(&sAh[r][c]) = vh;
      *(uint4*)(&sAl[r][c]) = vl;
    }
#pragma unroll
    for (int i = 0; i < 4; ++i) {
      int lin = tid + (i << 8);
      int n   = lin >> 3;                       // d row in tile
      int c   = (lin & 7) << 3;
      size_t o = (size_t)(e * DD + d0 + n) * HH + kc + c;
      *(uint4*)(&sBh[n][c]) = *(const uint4*)(w2h + o);
      *(uint4*)(&sBl[n][c]) = *(const uint4*)(w2l + o);
    }
    __syncthreads();

#pragma unroll
    for (int kk = 0; kk < 64; kk += 32) {
      v16bf a0h = ld_fragA(&sAh[mrow][0],      kk, aoff);
      v16bf a0l = ld_fragA(&sAl[mrow][0],      kk, aoff);
      v16bf a1h = ld_fragA(&sAh[16 + mrow][0], kk, aoff);
      v16bf a1l = ld_fragA(&sAl[16 + mrow][0], kk, aoff);
      v16bf bh  = ld_fragB(&sBh[nn][0], kk, boff);
      v16bf bl  = ld_fragB(&sBl[nn][0], kk, boff);
      acc0 = wmma_bf(a0l, bh, acc0);
      acc0 = wmma_bf(a0h, bl, acc0);
      acc0 = wmma_bf(a0h, bh, acc0);
      acc1 = wmma_bf(a1l, bh, acc1);
      acc1 = wmma_bf(a1h, bl, acc1);
      acc1 = wmma_bf(a1h, bh, acc1);
    }
  }

  const int mbase = half << 3;
#pragma unroll
  for (int vi = 0; vi < 8; ++vi) {
    int p0 = sPair[vi + mbase];
    if (p0 >= 0) out[(size_t)p0 * DD + d0 + nn] = acc0[vi];
    int p1 = sPair[16 + vi + mbase];
    if (p1 >= 0) out[(size_t)p1 * DD + d0 + nn] = acc1[vi];
  }
}

// ---------------------------------------------------------------------------
// Fallback fp32 path (V_WMMA_F32_16X16X4_F32), used when ws is small
// ---------------------------------------------------------------------------
__global__ __launch_bounds__(256)
void moe_gemm1_f32(const float* __restrict__ x, const float* __restrict__ w1,
                   const float* __restrict__ w3, const int* __restrict__ counts,
                   const int* __restrict__ lists, float* __restrict__ g) {
  const int e = blockIdx.z, cnt = counts[e], row0 = blockIdx.y * 16;
  if (row0 >= cnt) return;
  const int h0 = blockIdx.x * 128;
  __shared__ float sA[16][68], sB1[128][68], sB3[128][68];
  __shared__ int sPair[16];
  const int tid = threadIdx.x, wave = tid >> 5, lane = tid & 31;
  const int m = lane & 15, koff = (lane >> 4) << 1, nn = (wave << 4) + (lane & 15);
  if (tid < 16) { int r = row0 + tid; sPair[tid] = (r < cnt) ? lists[e * PAIRS + r] : -1; }
  __syncthreads();
  v8f acc1 = {}, acc3 = {};
  for (int kc = 0; kc < DD; kc += 64) {
    __syncthreads();
    { int r = tid >> 4, c4 = (tid & 15) << 2, p = sPair[r];
      float4 v = make_float4(0.f, 0.f, 0.f, 0.f);
      if (p >= 0) v = *(const float4*)(x + (size_t)(p >> 1) * DD + kc + c4);
      *(float4*)(&sA[r][c4]) = v; }
#pragma unroll
    for (int i = 0; i < 8; ++i) {
      int lin = tid + (i << 8), n = lin >> 4, c4 = (lin & 15) << 2;
      size_t off = (size_t)(e * HH + h0 + n) * DD + kc + c4;
      *(float4*)(&sB1[n][c4]) = *(const float4*)(w1 + off);
      *(float4*)(&sB3[n][c4]) = *(const float4*)(w3 + off);
    }
    __syncthreads();
#pragma unroll
    for (int kk = 0; kk < 64; kk += 4) {
      v2f a  = *(const v2f*)(&sA[m][kk + koff]);
      v2f b1 = *(const v2f*)(&sB1[nn][kk + koff]);
      v2f b3 = *(const v2f*)(&sB3[nn][kk + koff]);
      acc1 = wmma_f32(a, b1, acc1);
      acc3 = wmma_f32(a, b3, acc3);
    }
  }
  const int mbase = (lane >> 4) << 3;
#pragma unroll
  for (int vi = 0; vi < 8; ++vi) {
    int p = sPair[vi + mbase];
    if (p >= 0) {
      float a1 = acc1[vi], a3 = acc3[vi];
      g[(size_t)p * HH + h0 + nn] = (a1 / (1.0f + __expf(-a1))) * a3;
    }
  }
}

__global__ __launch_bounds__(256)
void moe_gemm2_f32(const float* __restrict__ g, const float* __restrict__ w2,
                   const int* __restrict__ counts, const int* __restrict__ lists,
                   float* __restrict__ out) {
  const int e = blockIdx.z, cnt = counts[e], row0 = blockIdx.y * 16;
  if (row0 >= cnt) return;
  const int d0 = blockIdx.x * 128;
  __shared__ float sA[16][68], sB[64][132];
  __shared__ int sPair[16];
  const int tid = threadIdx.x, wave = tid >> 5, lane = tid & 31;
  const int m = lane & 15, koff = (lane >> 4) << 1, nn = (wave << 4) + (lane & 15);
  if (tid < 16) { int r = row0 + tid; sPair[tid] = (r < cnt) ? lists[e * PAIRS + r] : -1; }
  __syncthreads();
  v8f acc = {};
  for (int kc = 0; kc < HH; kc += 64) {
    __syncthreads();
    { int r = tid >> 4, c4 = (tid & 15) << 2, p = sPair[r];
      float4 v = make_float4(0.f, 0.f, 0.f, 0.f);
      if (p >= 0) v = *(const float4*)(g + (size_t)p * HH + kc + c4);
      *(float4*)(&sA[r][c4]) = v; }
#pragma unroll
    for (int i = 0; i < 8; ++i) {
      int lin = tid + (i << 8), k = lin >> 5, c4 = (lin & 31) << 2;
      size_t off = (size_t)(e * HH + kc + k) * DD + d0 + c4;
      *(float4*)(&sB[k][c4]) = *(const float4*)(w2 + off);
    }
    __syncthreads();
#pragma unroll
    for (int kk = 0; kk < 64; kk += 4) {
      v2f a = *(const v2f*)(&sA[m][kk + koff]);
      v2f b; b.x = sB[kk + koff][nn]; b.y = sB[kk + koff + 1][nn];
      acc = wmma_f32(a, b, acc);
    }
  }
  const int mbase = (lane >> 4) << 3;
#pragma unroll
  for (int vi = 0; vi < 8; ++vi) {
    int p = sPair[vi + mbase];
    if (p >= 0) out[(size_t)p * DD + d0 + nn] = acc[vi];
  }
}

// ---------------------------------------------------------------------------
// Launch
// ---------------------------------------------------------------------------
extern "C" void kernel_launch(void* const* d_in, const int* in_sizes, int n_in,
                              void* d_out, int out_size, void* d_ws, size_t ws_size,
                              hipStream_t stream) {
  const float* x   = (const float*)d_in[0];
  const int*   idx = (const int*)  d_in[1];
  const float* w1  = (const float*)d_in[2];
  const float* w2  = (const float*)d_in[3];
  const float* w3  = (const float*)d_in[4];
  float*       out = (float*)d_out;

  char* ws = (char*)d_ws;
  size_t off = 0;
  auto alloc = [&](size_t bytes) {
    size_t o = off;
    off = (off + bytes + 255) & ~(size_t)255;
    return o;
  };
  const size_t EHD = (size_t)EE * HH * DD;            // 23.07M elems
  size_t oCnt = alloc(EE * sizeof(int));
  size_t oLst = alloc((size_t)EE * PAIRS * sizeof(int));
  size_t oXh  = alloc((size_t)TT * DD * 2);
  size_t oXl  = alloc((size_t)TT * DD * 2);
  size_t oW1h = alloc(EHD * 2), oW1l = alloc(EHD * 2);
  size_t oW3h = alloc(EHD * 2), oW3l = alloc(EHD * 2);
  size_t oW2h = alloc(EHD * 2), oW2l = alloc(EHD * 2);
  size_t oGh  = alloc((size_t)PAIRS * HH * 2);
  size_t oGl  = alloc((size_t)PAIRS * HH * 2);
  const size_t needFast = off;                        // ~316 MB

  int*   counts = (int*)(ws + oCnt);
  int*   lists  = (int*)(ws + oLst);

  hipMemsetAsync(counts, 0, EE * sizeof(int), stream);
  moe_route<<<(PAIRS + 255) / 256, 256, 0, stream>>>(idx, counts, lists);

  if (ws_size >= needFast) {
    u16 *xh  = (u16*)(ws + oXh),  *xl  = (u16*)(ws + oXl);
    u16 *w1h = (u16*)(ws + oW1h), *w1l = (u16*)(ws + oW1l);
    u16 *w3h = (u16*)(ws + oW3h), *w3l = (u16*)(ws + oW3l);
    u16 *w2h = (u16*)(ws + oW2h), *w2l = (u16*)(ws + oW2l);
    u16 *gh  = (u16*)(ws + oGh),  *gl  = (u16*)(ws + oGl);

    long nx4 = (long)TT * DD / 4, nw4 = (long)EHD / 4;
    conv_split<<<(unsigned)((nx4 + 255) / 256), 256, 0, stream>>>(x,  xh,  xl,  nx4);
    conv_split<<<(unsigned)((nw4 + 255) / 256), 256, 0, stream>>>(w1, w1h, w1l, nw4);
    conv_split<<<(unsigned)((nw4 + 255) / 256), 256, 0, stream>>>(w3, w3h, w3l, nw4);
    conv_w2t<<<dim3(DD / 32, HH / 32, EE), 256, 0, stream>>>(w2, w2h, w2l);

    dim3 g1(HH / 128, PAIRS / 32, EE);   // (22, 128, 8)
    moe_gemm1_bf<<<g1, 256, 0, stream>>>(xh, xl, w1h, w1l, w3h, w3l,
                                         counts, lists, gh, gl);
    dim3 g2(DD / 128, PAIRS / 32, EE);   // (8, 128, 8)
    moe_gemm2_bf<<<g2, 256, 0, stream>>>(gh, gl, w2h, w2l, counts, lists, out);
  } else {
    // fp32 fallback: counts, lists, then g fp32 (~46.3 MB total)
    float* gF = (float*)(ws + oXh);
    dim3 g1(HH / 128, PAIRS / 16, EE);
    moe_gemm1_f32<<<g1, 256, 0, stream>>>(x, w1, w3, counts, lists, gF);
    dim3 g2(DD / 128, PAIRS / 16, EE);
    moe_gemm2_f32<<<g2, 256, 0, stream>>>(gF, w2, counts, lists, out);
  }
}
